// FusedCustomLinear_22823456211399
// MI455X (gfx1250) — matrix operand order repaired and checked by reading, not compiled, not run on hardware
//
#include <hip/hip_runtime.h>
#include <hip/hip_fp16.h>

typedef _Float16 v8h  __attribute__((ext_vector_type(8)));
typedef _Float16 v16h __attribute__((ext_vector_type(16)));
typedef float    v8f  __attribute__((ext_vector_type(8)));

#define FFN  4096
#define EMB  1024

// ---------------------------------------------------------------------------
// Kernel 1: h = fp16( gelu_f32(x) ) * mask * (1/0.9)   (each element once)
// 8 fp16 per thread: one b128 load of x, 8 mask bytes, one b128 store.
// ---------------------------------------------------------------------------
__global__ __launch_bounds__(256) void gelu_dropout_kernel(
    const _Float16* __restrict__ x,
    const unsigned char* __restrict__ mask,
    _Float16* __restrict__ h,
    long long total8)
{
    long long i = (long long)blockIdx.x * blockDim.x + threadIdx.x;
    if (i >= total8) return;

    const long long base = i * 8;
    v8h xv = *(const v8h*)(x + base);

    // 8 mask bytes (JAX bool = 1 byte each)
    unsigned long long mv = *(const unsigned long long*)(mask + base);

    const _Float16 scale = (_Float16)(1.0f / 0.9f);
    v8h out;
#pragma unroll
    for (int j = 0; j < 8; ++j) {
        float xf = (float)xv[j];
        // exact-form GELU in fp32, like the reference (erf path)
        float g = 0.5f * xf * (1.0f + erff(xf * 0.70710678118654752f));
        _Float16 gh = (_Float16)g;                       // cast to fp16 first (matches ref)
        unsigned char m = (unsigned char)(mv >> (8 * j));
        out[j] = m ? (_Float16)(gh * scale) : (_Float16)0;
    }
    *(v8h*)(h + base) = out;
}

// ---------------------------------------------------------------------------
// Kernel 2: y = h @ W^T + bias using v_wmma_f32_16x16x32_f16.
// 256 threads = 8 waves; block tile 128(M) x 128(E); wave tile 32 x 64
// = 2x4 grid of 16x16 WMMA tiles. Fragments loaded straight from global
// (weight is 8MB and h is 64MB -> both L2-resident on the 192MB L2).
// K unrolled by 64 with two fragment sets (ping-pong) so every WMMA octet
// consumes registers whose loads were issued a half-iteration earlier.
// ---------------------------------------------------------------------------
__global__ __launch_bounds__(256) void wmma_gemm_bias_kernel(
    const _Float16* __restrict__ h,      // [N, FFN]
    const _Float16* __restrict__ w,      // [EMB, FFN]
    const _Float16* __restrict__ bias,   // [EMB]
    float* __restrict__ out,             // [N, EMB] (fp16 values widened)
    int Nrows)
{
    const int lane  = threadIdx.x & 31;
    const int wave  = threadIdx.x >> 5;
    const int lhalf = lane & 15;       // 0..15
    const int hi    = lane >> 4;       // 0 or 1

    const int rowBase = blockIdx.x * 128 + (wave & 3) * 32;   // M
    const int colBase = blockIdx.y * 128 + (wave >> 2) * 64;  // E

    v8f c[2][4] = {};

    // A-matrix 16x32 fp16 layout (ISA 7.12.2):
    //  lanes 0-15 : row = lane,    K = {0..7, 16..23}
    //  lanes 16-31: row = lane-16, K = {8..15, 24..31}
    const _Float16* aptr[2];
#pragma unroll
    for (int mt = 0; mt < 2; ++mt)
        aptr[mt] = h + (long long)(rowBase + mt * 16 + lhalf) * FFN + hi * 8;

    // B-matrix 32x16 fp16 layout: lane = column e; lanes 0-15 hold K={0..7, 16..23},
    // lanes 16-31 hold K={8..15, 24..31} (same K order as A).
    const _Float16* bptr[4];
#pragma unroll
    for (int et = 0; et < 4; ++et)
        bptr[et] = w + (long long)(colBase + et * 16 + lhalf) * FFN + hi * 8;

    auto loadFrags = [&](v16h (&a)[2], v16h (&b)[4], int k) {
#pragma unroll
        for (int mt = 0; mt < 2; ++mt) {
            v8h lo = *(const v8h*)(aptr[mt] + k);        // K offset +0
            v8h up = *(const v8h*)(aptr[mt] + k + 16);   // K offset +16
            ((v8h*)&a[mt])[0] = lo;
            ((v8h*)&a[mt])[1] = up;
        }
#pragma unroll
        for (int et = 0; et < 4; ++et) {
            ((v8h*)&b[et])[0] = *(const v8h*)(bptr[et] + k);
            ((v8h*)&b[et])[1] = *(const v8h*)(bptr[et] + k + 16);
        }
    };

    auto doWmma = [&](const v16h (&a)[2], const v16h (&b)[4]) {
#pragma unroll
        for (int mt = 0; mt < 2; ++mt)
#pragma unroll
            for (int et = 0; et < 4; ++et)
                c[mt][et] = __builtin_amdgcn_wmma_f32_16x16x32_f16(
                    /*neg_a=*/false, a[mt],
                    /*neg_b=*/false, b[et],
                    /*c_mod=*/(short)0, c[mt][et],
                    /*reuse_a=*/false, /*reuse_b=*/false);
    };

    v16h a0[2], b0[4], a1[2], b1[4];
    loadFrags(a0, b0, 0);                    // prologue: first half-step

    for (int k0 = 0; k0 < FFN; k0 += 64) {
        loadFrags(a1, b1, k0 + 32);          // issue loads for 2nd half-step

        if (k0 + 128 <= FFN) {               // stage next A chunk into WGP$
            __builtin_prefetch(aptr[0] + k0 + 64, 0, 3);
            __builtin_prefetch(aptr[1] + k0 + 64, 0, 3);
        }

        doWmma(a0, b0);                      // compute on already-landed regs

        if (k0 + 64 < FFN)
            loadFrags(a0, b0, k0 + 64);      // issue loads for next iteration

        doWmma(a1, b1);
    }

    // C/D 16x16 f32 layout: VGPR r, lane L -> row = r + 8*hi, col = L&15.
#pragma unroll
    for (int mt = 0; mt < 2; ++mt) {
#pragma unroll
        for (int et = 0; et < 4; ++et) {
            const int col = colBase + et * 16 + lhalf;
            const _Float16 bv = bias[col];
#pragma unroll
            for (int r = 0; r < 8; ++r) {
                const int row = rowBase + mt * 16 + hi * 8 + r;
                _Float16 y16 = (_Float16)c[mt][et][r] + bv;   // fp16 add like ref
                out[(long long)row * EMB + col] = (float)y16;
            }
        }
    }
}

// ---------------------------------------------------------------------------
extern "C" void kernel_launch(void* const* d_in, const int* in_sizes, int n_in,
                              void* d_out, int out_size, void* d_ws, size_t ws_size,
                              hipStream_t stream)
{
    const _Float16*      x    = (const _Float16*)d_in[0];      // [N, FFN] fp16
    const _Float16*      w    = (const _Float16*)d_in[1];      // [EMB, FFN] fp16
    const _Float16*      bias = (const _Float16*)d_in[2];      // [EMB] fp16
    const unsigned char* mask = (const unsigned char*)d_in[3]; // [N, FFN] bool
    float*               out  = (float*)d_out;

    const long long total = (long long)in_sizes[0];  // N * FFN
    const int Nrows = (int)(total / FFN);            // 8192

    _Float16* hbuf = (_Float16*)d_ws;                // N*FFN fp16 = 64 MB scratch

    // Pass 1: activation + dropout (each element exactly once)
    const long long total8 = total / 8;
    const int blocks1 = (int)((total8 + 255) / 256);
    gelu_dropout_kernel<<<blocks1, 256, 0, stream>>>(x, mask, hbuf, total8);

    // Pass 2: WMMA GEMM + bias
    dim3 grid(Nrows / 128, EMB / 128);   // 64 x 8 = 512 workgroups
    wmma_gemm_bias_kernel<<<grid, 256, 0, stream>>>(hbuf, w, bias, out, Nrows);
}
